// contrastive_loss_sup_40329742909552
// MI455X (gfx1250) — matrix operand order repaired and checked by reading, not compiled, not run on hardware
//
#include <hip/hip_runtime.h>

#define B_DIM 4
#define D_DIM 128
#define N_DIM 4096

constexpr float INV_T     = 1.0f / 0.07f;
constexpr float MSHIFT    = 1.0f / 0.07f;   // fixed lse shift: logits <= 1/T since |sim|<=1
constexpr float DIAG_FILL = -10.0f;
constexpr float L1_EPS    = 1e-12f;
constexpr float LOG2E     = 1.4426950408889634f;
constexpr float LN2       = 0.6931471805599453f;

// exp(x*INV_T - MSHIFT) == exp2(x*EK1 + EK0)
constexpr float EK1 = INV_T * LOG2E;
constexpr float EK0 = -MSHIFT * LOG2E;
constexpr float DIAG_ARG = (DIAG_FILL * INV_T - MSHIFT) * LOG2E;

typedef __attribute__((ext_vector_type(16))) __bf16        v16bf;
typedef __attribute__((ext_vector_type(8)))  float         v8f;
typedef __attribute__((ext_vector_type(4)))  unsigned int  u32x4;

union FragBF { v16bf v; u32x4 u[2]; };

__device__ inline unsigned int pack_bf16x2(float a, float b) {
  unsigned int ua = __float_as_uint(a);
  unsigned int ub = __float_as_uint(b);
  ua = (ua + 0x7FFFu + ((ua >> 16) & 1u)) >> 16;   // RNE to bf16
  ub = (ub + 0x7FFFu + ((ub >> 16) & 1u)) >> 16;
  return ua | (ub << 16);
}
__device__ inline float bf_lo(unsigned int u) { return __uint_as_float(u << 16); }
__device__ inline float bf_hi(unsigned int u) { return __uint_as_float(u & 0xffff0000u); }

// ---------------------------------------------------------------------------
// Kernel 1: L1-normalize along D and pack [B,D,N] f32 -> [B,N,D] bf16.
// ---------------------------------------------------------------------------
__global__ __launch_bounds__(256) void l1norm_pack_kernel(
    const float* __restrict__ src, __bf16* __restrict__ dst) {
  const int nb = blockIdx.x % (N_DIM / 32);
  const int b  = blockIdx.x / (N_DIM / 32);
  const int n0 = nb * 32;
  const int tx = threadIdx.x;
  const int ty = threadIdx.y;

  __shared__ float tile[D_DIM][33];
  __shared__ float red[8][32];
  __shared__ float inv[32];

  const float* sb = src + (size_t)b * D_DIM * N_DIM;
  float acc = 0.f;
  #pragma unroll
  for (int d = ty; d < D_DIM; d += 8) {
    float v = sb[(size_t)d * N_DIM + n0 + tx];
    tile[d][tx] = v;
    acc += fabsf(v);
  }
  red[ty][tx] = acc;
  __syncthreads();
  if (ty == 0) {
    float s = 0.f;
    #pragma unroll
    for (int i = 0; i < 8; ++i) s += red[i][tx];
    inv[tx] = 1.0f / fmaxf(s, L1_EPS);
  }
  __syncthreads();

  const int t  = ty * 32 + tx;
  const int nw = t >> 3;
  const int d0 = (t & 7) * 16;
  const float sc = inv[nw];

  u32x4 p0, p1;
  p0.x = pack_bf16x2(tile[d0 +  0][nw] * sc, tile[d0 +  1][nw] * sc);
  p0.y = pack_bf16x2(tile[d0 +  2][nw] * sc, tile[d0 +  3][nw] * sc);
  p0.z = pack_bf16x2(tile[d0 +  4][nw] * sc, tile[d0 +  5][nw] * sc);
  p0.w = pack_bf16x2(tile[d0 +  6][nw] * sc, tile[d0 +  7][nw] * sc);
  p1.x = pack_bf16x2(tile[d0 +  8][nw] * sc, tile[d0 +  9][nw] * sc);
  p1.y = pack_bf16x2(tile[d0 + 10][nw] * sc, tile[d0 + 11][nw] * sc);
  p1.z = pack_bf16x2(tile[d0 + 12][nw] * sc, tile[d0 + 13][nw] * sc);
  p1.w = pack_bf16x2(tile[d0 + 14][nw] * sc, tile[d0 + 15][nw] * sc);

  __bf16* drow = dst + ((size_t)b * N_DIM + n0 + nw) * D_DIM + d0;
  *(u32x4*)drow       = p0;
  *(u32x4*)(drow + 8) = p1;
}

// ---------------------------------------------------------------------------
// Kernel 2: fused GEMM + streaming logsumexp + softplus-mean.
// Workgroup = 4 waves, owns 32 rows (two 16-row WMMA blocks). Each wave
// streams 64 of the 256 column tiles; every B fragment feeds 2 WMMAs
// (halves K traffic vs 16-row waves). No branch in the hot loop: the
// diagonal term is corrected after the loop from a recomputed self-dot.
// ---------------------------------------------------------------------------
__global__ __launch_bounds__(128) void lse_loss_kernel(
    const __bf16* __restrict__ qn, const __bf16* __restrict__ kn,
    float* __restrict__ out) {
  const int wg = blockIdx.x % (N_DIM / 32);
  const int b  = blockIdx.x / (N_DIM / 32);
  const int r0 = wg * 32;

  const int  tid  = threadIdx.x;
  const int  wave = tid >> 5;
  const int  lane = tid & 31;
  const bool lo   = lane < 16;
  const int  m    = lane & 15;

  __shared__ float sdata[4][32];   // per-wave partial sums for 32 rows

  const __bf16* qb  = qn + (size_t)b * N_DIM * D_DIM;
  const __bf16* kbp = kn + (size_t)b * N_DIM * D_DIM;

  // A fragments (16x32 bf16 MxK): lanes 0-15 row m hold K=[kb..kb+7],[kb+16..kb+23];
  // lanes 16-31 hold K=[kb+8..kb+15],[kb+24..kb+31]. Two row blocks.
  FragBF a0[4], a1[4];
  #pragma unroll
  for (int t = 0; t < 4; ++t) {
    const __bf16* p0 = qb + (size_t)(r0 + m) * D_DIM + t * 32 + (lo ? 0 : 8);
    a0[t].u[0] = *(const u32x4*)p0;
    a0[t].u[1] = *(const u32x4*)(p0 + 16);
    const __bf16* p1 = qb + (size_t)(r0 + 16 + m) * D_DIM + t * 32 + (lo ? 0 : 8);
    a1[t].u[0] = *(const u32x4*)p1;
    a1[t].u[1] = *(const u32x4*)(p1 + 16);
  }

  float s0[8], s1[8];
  #pragma unroll
  for (int r = 0; r < 8; ++r) { s0[r] = 0.f; s1[r] = 0.f; }

  const int j0 = wave * (N_DIM / 16 / 4);   // 64 tiles per wave
  const int j1 = j0 + (N_DIM / 16 / 4);
  for (int j = j0; j < j1; ++j) {
    const int c0t = j * 16;

    // B fragments (32x16 bf16 KxN): lane holds column m; lanes 0-15 cover
    // K=[kb..kb+7],[kb+16..kb+23], lanes 16-31 cover K=[kb+8..kb+15],[kb+24..kb+31].
    FragBF bf[4];
    #pragma unroll
    for (int t = 0; t < 4; ++t) {
      const __bf16* p = kbp + (size_t)(c0t + m) * D_DIM + t * 32 + (lo ? 0 : 8);
      bf[t].u[0] = *(const u32x4*)p;
      bf[t].u[1] = *(const u32x4*)(p + 16);
    }
    if (j + 1 < j1)
      __builtin_prefetch(kbp + (size_t)(c0t + 16 + m) * D_DIM, 0, 3);

    v8f c0 = {0.f, 0.f, 0.f, 0.f, 0.f, 0.f, 0.f, 0.f};
    v8f c1 = {0.f, 0.f, 0.f, 0.f, 0.f, 0.f, 0.f, 0.f};
    #pragma unroll
    for (int t = 0; t < 4; ++t) {
      c0 = __builtin_amdgcn_wmma_f32_16x16x32_bf16(
               false, a0[t].v, false, bf[t].v, (short)0, c0, false, false);
      c1 = __builtin_amdgcn_wmma_f32_16x16x32_bf16(
               false, a1[t].v, false, bf[t].v, (short)0, c1, false, false);
    }

    #pragma unroll
    for (int r = 0; r < 8; ++r) {
      s0[r] += __builtin_amdgcn_exp2f(__builtin_fmaf(c0[r], EK1, EK0));
      s1[r] += __builtin_amdgcn_exp2f(__builtin_fmaf(c1[r], EK1, EK0));
    }
  }

  // Per-row sums across each 16-lane half (row = r + 8*hi [+16 for block 1]).
  float v0[8], v1[8];
  #pragma unroll
  for (int r = 0; r < 8; ++r) {
    float x = s0[r];
    x += __shfl_xor(x, 1); x += __shfl_xor(x, 2);
    x += __shfl_xor(x, 4); x += __shfl_xor(x, 8);
    v0[r] = x;
    float y = s1[r];
    y += __shfl_xor(y, 1); y += __shfl_xor(y, 2);
    y += __shfl_xor(y, 4); y += __shfl_xor(y, 8);
    v1[r] = y;
  }
  if (m == 0) {
    const int h = lo ? 0 : 8;
    #pragma unroll
    for (int r = 0; r < 8; ++r) {
      sdata[wave][r + h]      = v0[r];
      sdata[wave][16 + r + h] = v1[r];
    }
  }
  __syncthreads();

  if (wave == 0) {
    // combined sum over all 4096 columns for row (r0 + lane)
    float v = sdata[0][lane] + sdata[1][lane] + sdata[2][lane] + sdata[3][lane];

    // Recompute self-dot q[row]·k[row] in f32 from bf16 fragments (lane m
    // holds the K-subset of rows r0+m and r0+16+m; pair lanes m,m+16 cover K).
    float dp0 = 0.f, dp1 = 0.f;
    #pragma unroll
    for (int t = 0; t < 4; ++t) {
      const __bf16* p0 = kbp + (size_t)(r0 + m) * D_DIM + t * 32 + (lo ? 0 : 8);
      FragBF k0; k0.u[0] = *(const u32x4*)p0; k0.u[1] = *(const u32x4*)(p0 + 16);
      const __bf16* p1 = kbp + (size_t)(r0 + 16 + m) * D_DIM + t * 32 + (lo ? 0 : 8);
      FragBF k1; k1.u[0] = *(const u32x4*)p1; k1.u[1] = *(const u32x4*)(p1 + 16);
      #pragma unroll
      for (int h = 0; h < 2; ++h)
        #pragma unroll
        for (int w = 0; w < 4; ++w) {
          const unsigned ua0 = a0[t].u[h][w], uk0 = k0.u[h][w];
          const unsigned ua1 = a1[t].u[h][w], uk1 = k1.u[h][w];
          dp0 = __builtin_fmaf(bf_lo(ua0), bf_lo(uk0), dp0);
          dp0 = __builtin_fmaf(bf_hi(ua0), bf_hi(uk0), dp0);
          dp1 = __builtin_fmaf(bf_lo(ua1), bf_lo(uk1), dp1);
          dp1 = __builtin_fmaf(bf_hi(ua1), bf_hi(uk1), dp1);
        }
    }
    const float dot0 = dp0 + __shfl_xor(dp0, 16);   // row r0+m   (lanes m, m+16)
    const float dot1 = dp1 + __shfl_xor(dp1, 16);   // row r0+16+m
    const float dvlo = __shfl(dot0, m);
    const float dvhi = __shfl(dot1, m);
    const float dv   = lo ? dvlo : dvhi;            // self-dot of row r0+lane

    // replace the erroneous diagonal term with the DIAG_FILL term
    const float vc = v - __builtin_amdgcn_exp2f(__builtin_fmaf(dv, EK1, EK0))
                       + __builtin_amdgcn_exp2f(DIAG_ARG);

    const float lse  = MSHIFT + __builtin_amdgcn_logf(vc) * LN2;
    float loss = fmaxf(lse, 0.f) + log1pf(__expf(-fabsf(lse)));
    loss += __shfl_xor(loss, 1);  loss += __shfl_xor(loss, 2);
    loss += __shfl_xor(loss, 4);  loss += __shfl_xor(loss, 8);
    loss += __shfl_xor(loss, 16);
    if (lane == 0)
      atomicAdd(out, loss * (1.0f / (float)(B_DIM * N_DIM)));
  }
}

// ---------------------------------------------------------------------------
extern "C" void kernel_launch(void* const* d_in, const int* in_sizes, int n_in,
                              void* d_out, int out_size, void* d_ws, size_t ws_size,
                              hipStream_t stream) {
  (void)in_sizes; (void)n_in; (void)out_size; (void)ws_size;
  const float* fq = (const float*)d_in[0];
  const float* fk = (const float*)d_in[1];
  float* out = (float*)d_out;

  const size_t packed_bytes = (size_t)B_DIM * N_DIM * D_DIM * sizeof(__bf16); // 4 MB
  __bf16* qn = (__bf16*)d_ws;
  __bf16* kn = (__bf16*)((char*)d_ws + packed_bytes);

  hipMemsetAsync(out, 0, sizeof(float), stream);

  dim3 ngrid(B_DIM * (N_DIM / 32));
  dim3 nblk(32, 8);
  l1norm_pack_kernel<<<ngrid, nblk, 0, stream>>>(fq, qn);
  l1norm_pack_kernel<<<ngrid, nblk, 0, stream>>>(fk, kn);

  lse_loss_kernel<<<B_DIM * (N_DIM / 32), 128, 0, stream>>>(qn, kn, out);
}
